// Model_57157424775294
// MI455X (gfx1250) — matrix operand order repaired and checked by reading, not compiled, not run on hardware
//
#include <hip/hip_runtime.h>
#include <hip/hip_bf16.h>

#define B_ 32
#define T_ 32
#define S_ 512
#define EMB_ 300
#define ENC_ 512
#define DEC_ 512
#define ATT_ 512
#define VOCAB_ 30000
#define NUM_OOV_ 50
#define EXT_ (VOCAB_ + NUM_OOV_)
#define NEG_INF (-1.0e12f)

typedef __attribute__((ext_vector_type(16))) __bf16 v16bf;
typedef __attribute__((ext_vector_type(8)))  float  v8f;

// ---------------------------------------------------------------------------
// WMMA bf16 16x16x32 fragments, wave32; A and B both use contiguous k = 16h + s.
// A: M x K row-major bf16.  W: N x K row-major bf16 (so B = W^T).  C: f32.
// ---------------------------------------------------------------------------
static __device__ __forceinline__ v16bf load_a_frag(const __bf16* p, int half) {
  // lane half h holds K = {16h .. 16h+15} contiguously for row m = lane&15
  union { uint4 q[2]; v16bf v; } u;
  u.q[0] = *reinterpret_cast<const uint4*>(p + half * 16);
  u.q[1] = *reinterpret_cast<const uint4*>(p + half * 16 + 8);
  return u.v;
}
static __device__ __forceinline__ v16bf load_b_frag(const __bf16* p, int half) {
  // lane half h holds K = {16h .. 16h+15} contiguously for column n = lane&15
  union { uint4 q[2]; v16bf v; } u;
  const uint4* p4 = reinterpret_cast<const uint4*>(p + half * 16);
  u.q[0] = p4[0];
  u.q[1] = p4[1];
  return u.v;
}

// One wave computes a 32x16 macro-tile (2 M-tiles sharing each B fragment).
static __device__ __forceinline__ void wmma_tile_bf16_m2(
    const __bf16* A, int lda, const __bf16* W, int ldw,
    float* C, long ldc, const float* bias, int m0, int n0, int K) {
  const int lane = threadIdx.x & 31;
  const int half = lane >> 4;
  const int lr   = lane & 15;
  v8f acc0 = {}, acc1 = {};
  const __bf16* arow0 = A + (long)(m0 + lr) * lda;
  const __bf16* arow1 = A + (long)(m0 + 16 + lr) * lda;
  const __bf16* wrow  = W + (long)(n0 + lr) * ldw;
  for (int k0 = 0; k0 < K; k0 += 32) {
    v16bf b  = load_b_frag(wrow + k0, half);
    v16bf a0 = load_a_frag(arow0 + k0, half);
    v16bf a1 = load_a_frag(arow1 + k0, half);
    acc0 = __builtin_amdgcn_wmma_f32_16x16x32_bf16(false, a0, false, b,
                                                   (short)0, acc0, false, false);
    acc1 = __builtin_amdgcn_wmma_f32_16x16x32_bf16(false, a1, false, b,
                                                   (short)0, acc1, false, false);
  }
  const int n = n0 + lr;
  const float bv = bias ? bias[n] : 0.0f;
#pragma unroll
  for (int i = 0; i < 8; ++i) {
    const int m = m0 + i + 8 * half;
    C[(long)m * ldc + n]        = acc0[i] + bv;
    C[(long)(m + 16) * ldc + n] = acc1[i] + bv;
  }
}

// ---------------------------------------------------------------------------
// Utility kernels
// ---------------------------------------------------------------------------
__global__ void k_f32_to_bf16(const float* __restrict__ in, __bf16* __restrict__ out, int n) {
  for (int i = blockIdx.x * blockDim.x + threadIdx.x; i < n; i += gridDim.x * blockDim.x)
    out[i] = (__bf16)in[i];
}

__global__ void k_zero_u32(unsigned int* __restrict__ p, int n) {
  for (int i = blockIdx.x * blockDim.x + threadIdx.x; i < n; i += gridDim.x * blockDim.x)
    p[i] = 0u;
}

static __device__ __forceinline__ float sigmoidf(float x) {
  return 1.0f / (1.0f + __expf(-x));
}
static __device__ __forceinline__ unsigned int fmap(float x) {
  unsigned int b = __float_as_uint(x);
  return (b & 0x80000000u) ? ~b : (b | 0x80000000u);
}
static __device__ __forceinline__ float funmap(unsigned int u) {
  return (u >> 31) ? __uint_as_float(u ^ 0x80000000u) : __uint_as_float(~u);
}

// pre_hidden = enc_states @ W_trans^T + b_trans  -> h0 (layer0), h1 (layer1)
__global__ void k_init_hidden(const float* __restrict__ enc_states,
                              const float* __restrict__ Wt, const float* __restrict__ bt,
                              float* __restrict__ h0, float* __restrict__ h1) {
  int tid = blockIdx.x * blockDim.x + threadIdx.x;
  if (tid >= 2 * B_ * DEC_) return;
  int j = tid & 511, b = (tid >> 9) & 31, l = tid >> 14;
  const float* x = enc_states + (((long)l * B_ + b) << 9);
  const float* w = Wt + (long)j * ENC_;
  float acc = bt[j];
  for (int k = 0; k < ENC_; ++k) acc += x[k] * w[k];
  (l == 0 ? h0 : h1)[(b << 9) + j] = acc;
}

// Generic GRU cell: x = concat(xa[b] (na), xb[b] (nb));  h_out = GRU(x, h_in)
__global__ void k_gru(const float* __restrict__ xa, long sa, int na,
                      const float* __restrict__ xb, long sb, int nb,
                      const float* __restrict__ h_in,
                      const float* __restrict__ Wih, const float* __restrict__ Whh,
                      const float* __restrict__ bih, const float* __restrict__ bhh,
                      float* __restrict__ h_out) {
  int tid = blockIdx.x * blockDim.x + threadIdx.x;
  if (tid >= B_ * DEC_) return;
  int j = tid & 511, b = tid >> 9;
  const float* pa = xa + (long)b * sa;
  const float* pb = xb ? (xb + (long)b * sb) : nullptr;
  const float* hp = h_in + (b << 9);
  const int nin = na + nb;
  float gi[3], gh[3];
#pragma unroll
  for (int g = 0; g < 3; ++g) {
    const int row = j + DEC_ * g;
    const float* wi = Wih + (long)row * nin;
    float s = bih[row];
    for (int k = 0; k < na; ++k) s += pa[k] * wi[k];
    for (int k = 0; k < nb; ++k) s += pb[k] * wi[na + k];
    gi[g] = s;
    const float* wh = Whh + (long)row * DEC_;
    float t = bhh[row];
    for (int k = 0; k < DEC_; ++k) t += hp[k] * wh[k];
    gh[g] = t;
  }
  float r = sigmoidf(gi[0] + gh[0]);
  float z = sigmoidf(gi[1] + gh[1]);
  float n = tanhf(gi[2] + r * gh[2]);
  h_out[(b << 9) + j] = (1.0f - z) * n + z * hp[j];
}

// q = h1 @ Wq^T
__global__ void k_query(const float* __restrict__ h1, const float* __restrict__ Wq,
                        float* __restrict__ q) {
  int tid = blockIdx.x * blockDim.x + threadIdx.x;
  if (tid >= B_ * ATT_) return;
  int j = tid & 511, b = tid >> 9;
  const float* h = h1 + (b << 9);
  const float* w = Wq + (long)j * DEC_;
  float s = 0.0f;
  for (int k = 0; k < DEC_; ++k) s += h[k] * w[k];
  q[(b << 9) + j] = s;
}

// energies + coverage output + pointer scatter-max
__global__ void k_energy(const float* __restrict__ precomp, const float* __restrict__ q,
                         const float* __restrict__ cov, const float* __restrict__ wcov,
                         const float* __restrict__ vvec, const int* __restrict__ mask,
                         const int* __restrict__ ext_idx,
                         float* __restrict__ energy, float* __restrict__ covs_out,
                         unsigned int* __restrict__ ptrmax, int t) {
  int tid = blockIdx.x * blockDim.x + threadIdx.x;
  if (tid >= B_ * S_) return;
  int s = tid & 511, b = tid >> 9;
  const float* pre = precomp + (long)tid * ATT_;
  const float* qb  = q + (b << 9);
  float cv = cov[tid];
  float e = 0.0f;
  for (int a = 0; a < ATT_; ++a)
    e += tanhf(pre[a] + qb[a] + cv * wcov[a]) * vvec[a];
  float en = (mask[tid] == 0) ? NEG_INF : e;
  energy[tid] = en;
  covs_out[((long)b * T_ + t) * S_ + s] = cv;
  atomicMax(&ptrmax[(long)b * EXT_ + ext_idx[tid]], fmap(en));
}

// softmax over S, write attn output, coverage += attn   (one block per batch row)
__global__ void k_softmax(const float* __restrict__ energy, float* __restrict__ attn,
                          float* __restrict__ cov, float* __restrict__ attns_out, int t) {
  __shared__ float red[256];
  const int b = blockIdx.x, tid = threadIdx.x;
  const float* e = energy + (b << 9);
  float m = NEG_INF;
  for (int s = tid; s < S_; s += 256) m = fmaxf(m, e[s]);
  red[tid] = m; __syncthreads();
  for (int o = 128; o > 0; o >>= 1) { if (tid < o) red[tid] = fmaxf(red[tid], red[tid + o]); __syncthreads(); }
  m = red[0]; __syncthreads();
  float sum = 0.0f;
  for (int s = tid; s < S_; s += 256) {
    float x = __expf(e[s] - m);
    attn[(b << 9) + s] = x;
    sum += x;
  }
  red[tid] = sum; __syncthreads();
  for (int o = 128; o > 0; o >>= 1) { if (tid < o) red[tid] += red[tid + o]; __syncthreads(); }
  float inv = 1.0f / red[0];
  for (int s = tid; s < S_; s += 256) {
    float a = attn[(b << 9) + s] * inv;
    attn[(b << 9) + s] = a;
    attns_out[((long)b * T_ + t) * S_ + s] = a;
    cov[(b << 9) + s] += a;
  }
}

// context = attn @ encoder_outputs
__global__ void k_context(const float* __restrict__ attn, const float* __restrict__ enc,
                          float* __restrict__ ctx) {
  int tid = blockIdx.x * blockDim.x + threadIdx.x;
  if (tid >= B_ * ENC_) return;
  int h = tid & 511, b = tid >> 9;
  const float* a = attn + (b << 9);
  const float* e = enc + ((long)b * S_) * ENC_ + h;
  float s = 0.0f;
  for (int j = 0; j < S_; ++j) s += a[j] * e[(long)j * ENC_];
  ctx[(b << 9) + h] = s;
}

// readout + maxout -> bf16 A matrix for the logit WMMA GEMM
__global__ void k_readout(const float* __restrict__ trg, int t,
                          const float* __restrict__ h1, const float* __restrict__ ctx,
                          const float* __restrict__ Wr, const float* __restrict__ br,
                          __bf16* __restrict__ Abf) {
  int tid = blockIdx.x * blockDim.x + threadIdx.x;
  if (tid >= B_ * (DEC_ / 2)) return;
  int p = tid & 255, b = tid >> 8;
  const float* emb = trg + ((long)b * T_ + t) * EMB_;
  const float* h = h1 + (b << 9);
  const float* c = ctx + (b << 9);
  const int NIN = EMB_ + DEC_ + ENC_;  // 1324
  float mx = -3.4e38f;
#pragma unroll
  for (int jj = 0; jj < 2; ++jj) {
    const int j = 2 * p + jj;
    const float* w = Wr + (long)j * NIN;
    float s = br[j];
    for (int k = 0; k < EMB_; ++k) s += emb[k] * w[k];
    for (int k = 0; k < DEC_; ++k) s += h[k] * w[EMB_ + k];
    for (int k = 0; k < ENC_; ++k) s += c[k] * w[EMB_ + DEC_ + k];
    mx = fmaxf(mx, s);
  }
  Abf[(b << 8) + p] = (__bf16)mx;
}

// precompute = encoder_outputs @ Wpre^T + bpre  (WMMA, M=16384 N=512 K=512)
// one wave per 32x16 macro-tile: 512 M-macro-tiles x 32 N-tiles = 16384 waves
__global__ void k_gemm_precompute(const __bf16* __restrict__ A, const __bf16* __restrict__ W,
                                  const float* __restrict__ bias, float* __restrict__ C) {
  int wid = (blockIdx.x * blockDim.x + threadIdx.x) >> 5;
  if (wid >= (B_ * S_ / 32) * (ATT_ / 16)) return;
  int mt = wid >> 5, nt = wid & 31;
  wmma_tile_bf16_m2(A, ENC_, W, ENC_, C, ATT_, bias, mt * 32, nt * 16, ENC_);
}

// logit = maxout @ W_logit^T + b_logit  (WMMA, M=32 N=30000 K=256), into d_out row t
// one wave per 32x16 macro-tile: each B fragment of the W_logit stream read once
__global__ void k_gemm_logit(const __bf16* __restrict__ A, const __bf16* __restrict__ W,
                             const float* __restrict__ bias, float* __restrict__ Cbase) {
  int wid = (blockIdx.x * blockDim.x + threadIdx.x) >> 5;
  const int NT = VOCAB_ / 16;                        // 1875
  if (wid >= NT) return;
  wmma_tile_bf16_m2(A, DEC_ / 2, W, DEC_ / 2, Cbase, (long)T_ * EXT_, bias,
                    0, wid * 16, DEC_ / 2);
}

// pointer-generator merge into d_out
__global__ void k_finalize(float* __restrict__ out, const unsigned int* __restrict__ ptrmax,
                           int t) {
  int tid = blockIdx.x * blockDim.x + threadIdx.x;
  if (tid >= B_ * EXT_) return;
  int b = tid / EXT_;
  int v = tid - b * EXT_;
  float* orow = out + ((long)b * T_ + t) * EXT_;
  float base = (v < VOCAB_) ? orow[v] : 0.0f;        // extended = [logit+bias, zeros]
  unsigned int u = ptrmax[(long)b * EXT_ + v];
  float ptr = 0.0f;
  if (u != 0u) {
    float d = funmap(u);
    ptr = (d == NEG_INF) ? 0.0f : d;                 // where(out==-INF, 0, out)
  }
  float val = base + ptr;
  if (val == 0.0f) val = NEG_INF;                    // where(logit_ext==0, -INF)
  orow[v] = val;
}

// ---------------------------------------------------------------------------
extern "C" void kernel_launch(void* const* d_in, const int* in_sizes, int n_in,
                              void* d_out, int out_size, void* d_ws, size_t ws_size,
                              hipStream_t stream) {
  const float* trg      = (const float*)d_in[0];
  const float* enc_st   = (const float*)d_in[1];
  const float* enc_out  = (const float*)d_in[2];
  const float* W_trans  = (const float*)d_in[3];
  const float* b_trans  = (const float*)d_in[4];
  const float* g0_Wih   = (const float*)d_in[5];
  const float* g0_Whh   = (const float*)d_in[6];
  const float* g0_bih   = (const float*)d_in[7];
  const float* g0_bhh   = (const float*)d_in[8];
  const float* g1_Wih   = (const float*)d_in[9];
  const float* g1_Whh   = (const float*)d_in[10];
  const float* g1_bih   = (const float*)d_in[11];
  const float* g1_bhh   = (const float*)d_in[12];
  const float* Wpre     = (const float*)d_in[13];
  const float* bpre     = (const float*)d_in[14];
  const float* Wq       = (const float*)d_in[15];
  const float* Wcov     = (const float*)d_in[16];
  const float* vvec     = (const float*)d_in[17];
  const float* W_read   = (const float*)d_in[18];
  const float* b_read   = (const float*)d_in[19];
  const float* W_logit  = (const float*)d_in[20];
  const float* b_logit  = (const float*)d_in[21];
  const int*   ext_idx  = (const int*)d_in[22];
  const int*   mask     = (const int*)d_in[23];

  float* out = (float*)d_out;
  float* attns_out = out + (long)B_ * T_ * EXT_;
  float* covs_out  = attns_out + (long)B_ * T_ * S_;

  // workspace carving (256B aligned regions)
  char* wp = (char*)d_ws;
  auto carve = [&](size_t bytes) -> void* {
    void* p = wp;
    wp += (bytes + 255) & ~(size_t)255;
    return p;
  };
  __bf16* wl_bf   = (__bf16*)carve((size_t)VOCAB_ * (DEC_ / 2) * 2);  // 15.36 MB
  __bf16* enc_bf  = (__bf16*)carve((size_t)B_ * S_ * ENC_ * 2);       // 16.78 MB
  __bf16* wpre_bf = (__bf16*)carve((size_t)ATT_ * ENC_ * 2);          // 0.52 MB
  float*  precomp = (float*) carve((size_t)B_ * S_ * ATT_ * 4);       // 33.55 MB
  unsigned int* ptrmax = (unsigned int*)carve((size_t)B_ * EXT_ * 4); // 3.85 MB
  float* h0A = (float*)carve(B_ * DEC_ * 4);
  float* h0B = (float*)carve(B_ * DEC_ * 4);
  float* h1A = (float*)carve(B_ * DEC_ * 4);
  float* h1B = (float*)carve(B_ * DEC_ * 4);
  float* ctx = (float*)carve(B_ * ENC_ * 4);
  float* cov = (float*)carve(B_ * S_ * 4);
  float* q   = (float*)carve(B_ * ATT_ * 4);
  float* ene = (float*)carve(B_ * S_ * 4);
  float* att = (float*)carve(B_ * S_ * 4);
  __bf16* Abf = (__bf16*)carve(B_ * (DEC_ / 2) * 2);

  const int TPB = 256;

  // ---- setup (recomputed every call; deterministic) ----
  {
    int n = VOCAB_ * (DEC_ / 2);
    k_f32_to_bf16<<<(n + TPB - 1) / TPB, TPB, 0, stream>>>(W_logit, wl_bf, n);
    n = B_ * S_ * ENC_;
    k_f32_to_bf16<<<(n + TPB - 1) / TPB, TPB, 0, stream>>>(enc_out, enc_bf, n);
    n = ATT_ * ENC_;
    k_f32_to_bf16<<<(n + TPB - 1) / TPB, TPB, 0, stream>>>(Wpre, wpre_bf, n);
    k_init_hidden<<<(2 * B_ * DEC_ + TPB - 1) / TPB, TPB, 0, stream>>>(
        enc_st, W_trans, b_trans, h0A, h1A);
    n = B_ * ENC_;
    k_zero_u32<<<(n + TPB - 1) / TPB, TPB, 0, stream>>>((unsigned int*)ctx, n);
    n = B_ * S_;
    k_zero_u32<<<(n + TPB - 1) / TPB, TPB, 0, stream>>>((unsigned int*)cov, n);
    int waves = (B_ * S_ / 32) * (ATT_ / 16);                 // 16384 waves
    k_gemm_precompute<<<(waves * 32 + TPB - 1) / TPB, TPB, 0, stream>>>(
        enc_bf, wpre_bf, bpre, precomp);
  }

  // ---- sequential scan over T steps ----
  for (int t = 0; t < T_; ++t) {
    float* h0_in  = (t & 1) ? h0B : h0A;
    float* h0_out = (t & 1) ? h0A : h0B;
    float* h1_in  = (t & 1) ? h1B : h1A;
    float* h1_out = (t & 1) ? h1A : h1B;

    // GRU layer 0: x = [emb_t, ctx]
    k_gru<<<(B_ * DEC_ + TPB - 1) / TPB, TPB, 0, stream>>>(
        trg + (long)t * EMB_, (long)T_ * EMB_, EMB_,
        ctx, ENC_, ENC_, h0_in,
        g0_Wih, g0_Whh, g0_bih, g0_bhh, h0_out);
    // GRU layer 1: x = h0
    k_gru<<<(B_ * DEC_ + TPB - 1) / TPB, TPB, 0, stream>>>(
        h0_out, DEC_, DEC_, nullptr, 0, 0, h1_in,
        g1_Wih, g1_Whh, g1_bih, g1_bhh, h1_out);
    // attention query
    k_query<<<(B_ * ATT_ + TPB - 1) / TPB, TPB, 0, stream>>>(h1_out, Wq, q);
    // reset pointer scatter buffer (0 == "below every mapped energy")
    {
      int n = B_ * EXT_;
      k_zero_u32<<<(n + TPB - 1) / TPB, TPB, 0, stream>>>(ptrmax, n);
    }
    // energies + covs output + pointer scatter-max
    k_energy<<<(B_ * S_ + TPB - 1) / TPB, TPB, 0, stream>>>(
        precomp, q, cov, Wcov, vvec, mask, ext_idx, ene, covs_out, ptrmax, t);
    // softmax + attns output + coverage update
    k_softmax<<<B_, 256, 0, stream>>>(ene, att, cov, attns_out, t);
    // context vector
    k_context<<<(B_ * ENC_ + TPB - 1) / TPB, TPB, 0, stream>>>(att, enc_out, ctx);
    // readout + maxout -> bf16 A
    k_readout<<<(B_ * (DEC_ / 2) + TPB - 1) / TPB, TPB, 0, stream>>>(
        trg, t, h1_out, ctx, W_read, b_read, Abf);
    // big WMMA GEMM: logits (+bias) straight into d_out at step t
    {
      int waves = VOCAB_ / 16;                                 // 1875 waves
      k_gemm_logit<<<(waves * 32 + TPB - 1) / TPB, TPB, 0, stream>>>(
          Abf, wl_bf, b_logit, out + (long)t * EXT_);
    }
    // pointer-generator merge
    {
      int n = B_ * EXT_;
      k_finalize<<<(n + TPB - 1) / TPB, TPB, 0, stream>>>(out, ptrmax, t);
    }
  }
}
